// SparseCINConv_89163521065166
// MI455X (gfx1250) — matrix operand order repaired and checked by reading, not compiled, not run on hardware
//
#include <hip/hip_runtime.h>
#include <hip/hip_bf16.h>

// ---------------------------------------------------------------------------
// CDNA5 (gfx1250) fused Sparse-CIN forward.
//  - segment sums via global_atomic_add_f32 scatter
//  - all dense math via v_wmma_f32_16x16x32_bf16 (wave32 WMMA)
//  - BN stats accumulated atomically in the GEMM epilogue
//  - BN apply + ReLU fused into the next GEMM's A-operand load (bf16 convert)
// ---------------------------------------------------------------------------

typedef __attribute__((ext_vector_type(16))) __bf16 v16bf;
typedef __attribute__((ext_vector_type(8)))  float  v8f;

__device__ __forceinline__ unsigned short f2bf(float f) {
  unsigned int x = __float_as_uint(f);
  unsigned int r = x + 0x7FFFu + ((x >> 16) & 1u);   // round-to-nearest-even
  return (unsigned short)(r >> 16);
}

// Per-source descriptor for the fused A-operand transform.
struct Src {
  const float* base;   // may be null -> 0
  const float* add;    // may be null; contributes (1+eps)*add
  const float* eps;    // device ptr to eps vector (or null -> factor 1)
  int          epsIdx;
  const float* scale;  // per-column BN scale (null -> identity)
  const float* shift;  // per-column BN shift
  int          relu;   // apply ReLU after scale/shift
};

__device__ __forceinline__ float load_src(const Src& S, int row, int k) {
  float v = 0.0f;
  if (S.base) v = S.base[(size_t)row * 64 + k];
  if (S.add) {
    float f = 1.0f + (S.eps ? S.eps[S.epsIdx] : 0.0f);
    v += f * S.add[(size_t)row * 64 + k];
  }
  if (S.scale) v = v * S.scale[k] + S.shift[k];
  if (S.relu)  v = fmaxf(v, 0.0f);
  return v;
}

// GEMM:  out[N x 64] = transform(A)[N x KC] @ W[KC x 64] + bias,
// accumulating per-column sum / sum-of-squares for BatchNorm.
// Block: 128 threads = 4 wave32; tile 64 rows x 64 cols.
template <int KC>
__global__ void gemm_bn_kernel(int Nrows, Src s0, Src s1,
                               const unsigned short* __restrict__ WT, // [64][KC] bf16 (N-major)
                               const float* __restrict__ bias,
                               float* __restrict__ outPre,
                               float* __restrict__ colSum,
                               float* __restrict__ colSq) {
  constexpr int STRIDE = KC + 8;
  __shared__ unsigned short sA[64 * STRIDE];

  const int tid = threadIdx.x;
  const int blockRow = blockIdx.x * 64;

  // ---- stage A tile to LDS as bf16 (fused transform), uniform trip count ----
  for (int idx = tid; idx < 64 * KC; idx += 128) {
    int r = idx / KC;
    int k = idx - r * KC;
    int rg = blockRow + r;
    float v = 0.0f;
    if (rg < Nrows) {
      if (KC == 128 && k >= 64) v = load_src(s1, rg, k - 64);
      else                      v = load_src(s0, rg, k);
    }
    sA[r * STRIDE + k] = f2bf(v);
  }
  __syncthreads();

  const int w   = tid >> 5;          // wave id 0..3 -> 16-row strip
  const int ln  = tid & 31;
  const int hlf = ln >> 4;           // lane half
  const int lm  = ln & 15;
  const int rl  = w * 16 + lm;       // A row within tile for this lane

  v8f acc[4] = {};

  for (int kb = 0; kb < KC; kb += 32) {
    // A fragment: contiguous k order, same as B; lane half selects K base (0/16),
    // element pairs p: K = kb + 16*half + 2p.
    union { unsigned int u[8]; v16bf v; } af;
    const int kbase = kb + hlf * 16;
#pragma unroll
    for (int p = 0; p < 8; ++p) {
      int K = kbase + 2 * p;
      af.u[p] = *(const unsigned int*)&sA[rl * STRIDE + K];
    }
#pragma unroll
    for (int nt = 0; nt < 4; ++nt) {
      int n = nt * 16 + lm;
      // B fragment: 32x16 bf16; lane half covers K kb..kb+15 / kb+16..kb+31,
      // contiguous in the N-major transposed weight -> one 32B load.
      const v16bf bf = *(const v16bf*)(const void*)(WT + (size_t)n * KC + kb + hlf * 16);
      acc[nt] = __builtin_amdgcn_wmma_f32_16x16x32_bf16(
          false, af.v, false, bf, (short)0, acc[nt], false, false);
    }
  }

  // ---- epilogue: bias, store pre-BN, accumulate column stats ----
#pragma unroll
  for (int nt = 0; nt < 4; ++nt) {
    int col = nt * 16 + lm;
    float b = bias[col];
    float lsum = 0.0f, lsq = 0.0f;
    int rowBase = blockRow + w * 16 + hlf * 8;
#pragma unroll
    for (int j = 0; j < 8; ++j) {
      int rg = rowBase + j;
      if (rg < Nrows) {
        float v = acc[nt][j] + b;
        outPre[(size_t)rg * 64 + col] = v;
        lsum += v;
        lsq  += v * v;
      }
    }
    atomicAdd(&colSum[col], lsum);
    atomicAdd(&colSq[col],  lsq);
  }
}

// mean/var -> per-column scale/shift (BN training mode, biased variance).
__global__ void bn_finalize_kernel(const float* __restrict__ sum,
                                   const float* __restrict__ sq, int Nrows,
                                   const float* __restrict__ g,
                                   const float* __restrict__ be,
                                   float* __restrict__ scale,
                                   float* __restrict__ shift) {
  int c = threadIdx.x;
  float inv = 1.0f / (float)Nrows;
  float m   = sum[c] * inv;
  float var = sq[c] * inv - m * m;
  float sc  = g[c] * rsqrtf(var + 1e-5f);
  scale[c] = sc;
  shift[c] = be[c] - m * sc;
}

// final BN+ReLU into output
__global__ void bn_relu_store_kernel(const float* __restrict__ pre,
                                     const float* __restrict__ scale,
                                     const float* __restrict__ shift,
                                     float* __restrict__ out, int Nrows) {
  size_t i = (size_t)blockIdx.x * blockDim.x + threadIdx.x;
  size_t n = (size_t)Nrows * 64;
  if (i < n) {
    int c = (int)(i & 63);
    out[i] = fmaxf(pre[i] * scale[c] + shift[c], 0.0f);
  }
}

// segment-sum scatter: out[dst[e]] += x[src[e]]  (64 ch, float4 per thread)
__global__ void scatter_add_kernel(const float* __restrict__ x,
                                   const int* __restrict__ src,
                                   const int* __restrict__ dst,
                                   int E, float* __restrict__ out) {
  size_t t = (size_t)blockIdx.x * blockDim.x + threadIdx.x;
  if (t >= (size_t)E * 16) return;
  int e = (int)(t >> 4);
  int c = (int)(t & 15) * 4;
  int s = src[e], d = dst[e];
  const float4 v = *(const float4*)(x + (size_t)s * 64 + c);
  float* o = out + (size_t)d * 64 + c;
  atomicAdd(o + 0, v.x);
  atomicAdd(o + 1, v.y);
  atomicAdd(o + 2, v.z);
  atomicAdd(o + 3, v.w);
}

__global__ void zero_kernel(float* p, size_t n) {
  size_t i = (size_t)blockIdx.x * blockDim.x + threadIdx.x;
  if (i < n) p[i] = 0.0f;
}

// W [KC x 64] f32 row-major  ->  WT [64 x KC] bf16 (N-major, K contiguous)
__global__ void wconv_kernel(const float* __restrict__ W,
                             unsigned short* __restrict__ WT, int KC) {
  int i = blockIdx.x * blockDim.x + threadIdx.x;
  if (i >= 64 * KC) return;
  int n = i / KC;
  int k = i - n * KC;
  WT[n * KC + k] = f2bf(W[(size_t)k * 64 + n]);
}

// ---------------------------------------------------------------------------

extern "C" void kernel_launch(void* const* d_in, const int* in_sizes, int n_in,
                              void* d_out, int out_size, void* d_ws, size_t ws_size,
                              hipStream_t stream) {
  (void)n_in; (void)out_size; (void)ws_size;
  const float* x0 = (const float*)d_in[0];
  const float* x1 = (const float*)d_in[1];
  const float* x2 = (const float*)d_in[2];
  const int* up0  = (const int*)d_in[3];
  const int* up1  = (const int*)d_in[4];
  const int* bsrc1 = (const int*)d_in[5];
  const int* bdst1 = (const int*)d_in[6];
  const int* bsrc2 = (const int*)d_in[7];
  const int* bdst2 = (const int*)d_in[8];
  const float* up_W1 = (const float*)d_in[9];
  const float* up_b1 = (const float*)d_in[10];
  const float* up_g1 = (const float*)d_in[11];
  const float* up_be1 = (const float*)d_in[12];
  const float* up_W2 = (const float*)d_in[13];
  const float* up_b2 = (const float*)d_in[14];
  const float* up_g2 = (const float*)d_in[15];
  const float* up_be2 = (const float*)d_in[16];
  const float* bd_W1 = (const float*)d_in[17];
  const float* bd_b1 = (const float*)d_in[18];
  const float* bd_g1 = (const float*)d_in[19];
  const float* bd_be1 = (const float*)d_in[20];
  const float* bd_W2 = (const float*)d_in[21];
  const float* bd_b2 = (const float*)d_in[22];
  const float* bd_g2 = (const float*)d_in[23];
  const float* bd_be2 = (const float*)d_in[24];
  const float* cmb_W = (const float*)d_in[25];
  const float* cmb_b = (const float*)d_in[26];
  const float* cmb_g = (const float*)d_in[27];
  const float* cmb_be = (const float*)d_in[28];
  const float* epsP  = (const float*)d_in[29];

  const int N0 = in_sizes[0] / 64, N1 = in_sizes[1] / 64, N2 = in_sizes[2] / 64;
  const int EU0 = in_sizes[3] / 2, EU1 = in_sizes[4] / 2;
  const int EB1 = in_sizes[5], EB2 = in_sizes[7];
  const int Nmax = max(N0, max(N1, N2));

  // ---- workspace layout ----
  unsigned short* WTb = (unsigned short*)d_ws;        // bf16 transposed weights
  // per-d offsets (elements)
  const size_t oUW1 = 0, oUW2 = 3 * 4096, oBW1 = 6 * 4096, oBW2 = 9 * 4096;
  const size_t oCMB = 12 * 4096;                       // 3 * 8192 after here
  const size_t wtBytes = (12 * 4096 + 3 * 8192) * 2;   // 147456
  float* stats = (float*)((char*)d_ws + wtBytes);      // 5 stages * 256 floats
  const size_t statsBytes = 5 * 256 * sizeof(float);
  char* big = (char*)d_ws + ((wtBytes + statsBytes + 255) & ~(size_t)255);
  const size_t S = (size_t)Nmax * 64 * sizeof(float);
  float* aggU = (float*)(big);
  float* aggB = (float*)(big + S);
  float* H1   = (float*)(big + 2 * S);   // also reused as concat pre-BN
  float* H2u  = (float*)(big + 3 * S);
  float* H2b  = (float*)(big + 4 * S);

  float* outF = (float*)d_out;
  float* outPtr[3] = { outF, outF + (size_t)N0 * 64, outF + (size_t)N0 * 64 + (size_t)N1 * 64 };

  // ---- weight conversion (cheap, deterministic every call) ----
  for (int d = 0; d < 3; ++d) {
    int g64 = (64 * 64 + 255) / 256;
    wconv_kernel<<<g64, 256, 0, stream>>>(up_W1 + (size_t)d * 4096, WTb + oUW1 + d * 4096, 64);
    wconv_kernel<<<g64, 256, 0, stream>>>(up_W2 + (size_t)d * 4096, WTb + oUW2 + d * 4096, 64);
    wconv_kernel<<<g64, 256, 0, stream>>>(bd_W1 + (size_t)d * 4096, WTb + oBW1 + d * 4096, 64);
    wconv_kernel<<<g64, 256, 0, stream>>>(bd_W2 + (size_t)d * 4096, WTb + oBW2 + d * 4096, 64);
    int g128 = (64 * 128 + 255) / 256;
    wconv_kernel<<<g128, 256, 0, stream>>>(cmb_W + (size_t)d * 8192, WTb + oCMB + d * 8192, 128);
  }

  auto zero = [&](float* p, size_t n) {
    zero_kernel<<<(unsigned)((n + 255) / 256), 256, 0, stream>>>(p, n);
  };
  auto scat = [&](const float* x, const int* s, const int* d, int E, float* o) {
    size_t t = (size_t)E * 16;
    scatter_add_kernel<<<(unsigned)((t + 255) / 256), 256, 0, stream>>>(x, s, d, E, o);
  };

  auto runLevel = [&](int d, int Nr, const float* x, const float* aU, const float* aB, float* out) {
    zero(stats, 5 * 256);
    dim3 grid((Nr + 63) / 64);
    float* sum[5], *sq[5], *sc[5], *sh[5];
    for (int i = 0; i < 5; ++i) {
      sum[i] = stats + i * 256; sq[i] = sum[i] + 64; sc[i] = sum[i] + 128; sh[i] = sum[i] + 192;
    }
    Src dummy = {nullptr, nullptr, nullptr, 0, nullptr, nullptr, 0};

    // up branch
    Src a = {aU, x, epsP, d, nullptr, nullptr, 0};
    gemm_bn_kernel<64><<<grid, 128, 0, stream>>>(Nr, a, dummy, WTb + oUW1 + d * 4096,
                                                 up_b1 + d * 64, H1, sum[0], sq[0]);
    bn_finalize_kernel<<<1, 64, 0, stream>>>(sum[0], sq[0], Nr, up_g1 + d * 64, up_be1 + d * 64, sc[0], sh[0]);
    Src h = {H1, nullptr, nullptr, 0, sc[0], sh[0], 1};
    gemm_bn_kernel<64><<<grid, 128, 0, stream>>>(Nr, h, dummy, WTb + oUW2 + d * 4096,
                                                 up_b2 + d * 64, H2u, sum[1], sq[1]);
    bn_finalize_kernel<<<1, 64, 0, stream>>>(sum[1], sq[1], Nr, up_g2 + d * 64, up_be2 + d * 64, sc[1], sh[1]);

    // bd branch
    Src ab = {aB, x, epsP, d, nullptr, nullptr, 0};
    gemm_bn_kernel<64><<<grid, 128, 0, stream>>>(Nr, ab, dummy, WTb + oBW1 + d * 4096,
                                                 bd_b1 + d * 64, H1, sum[2], sq[2]);
    bn_finalize_kernel<<<1, 64, 0, stream>>>(sum[2], sq[2], Nr, bd_g1 + d * 64, bd_be1 + d * 64, sc[2], sh[2]);
    Src hb = {H1, nullptr, nullptr, 0, sc[2], sh[2], 1};
    gemm_bn_kernel<64><<<grid, 128, 0, stream>>>(Nr, hb, dummy, WTb + oBW2 + d * 4096,
                                                 bd_b2 + d * 64, H2b, sum[3], sq[3]);
    bn_finalize_kernel<<<1, 64, 0, stream>>>(sum[3], sq[3], Nr, bd_g2 + d * 64, bd_be2 + d * 64, sc[3], sh[3]);

    // concat GEMM (K=128): fused BN+ReLU of both branch outputs in the A-load
    Src cu = {H2u, nullptr, nullptr, 0, sc[1], sh[1], 1};
    Src cb = {H2b, nullptr, nullptr, 0, sc[3], sh[3], 1};
    gemm_bn_kernel<128><<<grid, 128, 0, stream>>>(Nr, cu, cb, WTb + oCMB + d * 8192,
                                                  cmb_b + d * 64, H1, sum[4], sq[4]);
    bn_finalize_kernel<<<1, 64, 0, stream>>>(sum[4], sq[4], Nr, cmb_g + d * 64, cmb_be + d * 64, sc[4], sh[4]);
    size_t tot = (size_t)Nr * 64;
    bn_relu_store_kernel<<<(unsigned)((tot + 255) / 256), 256, 0, stream>>>(H1, sc[4], sh[4], out, Nr);
  };

  // level 0: upper adjacency only
  zero(aggU, (size_t)N0 * 64);
  scat(x0, up0, up0 + EU0, EU0, aggU);
  runLevel(0, N0, x0, aggU, nullptr, outPtr[0]);

  // level 1: upper adjacency + node boundaries
  zero(aggU, (size_t)N1 * 64);
  scat(x1, up1, up1 + EU1, EU1, aggU);
  zero(aggB, (size_t)N1 * 64);
  scat(x0, bsrc1, bdst1, EB1, aggB);
  runLevel(1, N1, x1, aggU, aggB, outPtr[1]);

  // level 2: edge boundaries only
  zero(aggB, (size_t)N2 * 64);
  scat(x1, bsrc2, bdst2, EB2, aggB);
  runLevel(2, N2, x2, nullptr, aggB, outPtr[2]);
}
